// GraphNetworkKeras_41652592837404
// MI455X (gfx1250) — compile-verified, matrix operand order checked by reading, not run on hardware
//
#include <hip/hip_runtime.h>
#include <math.h>

// ---------------------------------------------------------------------------
// GraphNets (3 passes) for MI455X / gfx1250, wave32 + WMMA.
//
// d_in order: 0 nodes(100000x32) 1 edges(800000x32) 2 globals(64x16)
//             3 senders 4 receivers 5 node_gid 6 edge_gid
//             7..12  e_w1,e_b1,e_w2,e_b2,e_w3,e_b3   (112->16->16->32)
//             13..18 n_w1..n_b3                      (80->16->16->32)
//             19..24 g_w1..g_b3                      (80->16->16->16)
// d_out = [nodes | edges | globals] flat f32.
// ---------------------------------------------------------------------------

#define NNODES 100000
#define NEDGES 800000
#define NGRAPH 64
#define NPASS  3

typedef _Float16 half_t;
typedef __attribute__((ext_vector_type(16))) _Float16 v16h;
typedef __attribute__((ext_vector_type(8)))  float    v8f;

// ---------------------------------------------------------------------------
__global__ void zero_kernel(float* __restrict__ p, int n) {
  int i = blockIdx.x * blockDim.x + threadIdx.x;
  if (i < n) p[i] = 0.f;
}

__global__ void count_edges_kernel(const int* __restrict__ receivers,
                                   const int* __restrict__ edge_gid,
                                   float* __restrict__ node_count,
                                   float* __restrict__ gE_count) {
  int e = blockIdx.x * blockDim.x + threadIdx.x;
  if (e < NEDGES) {
    atomicAdd(&node_count[receivers[e]], 1.f);
    atomicAdd(&gE_count[edge_gid[e]], 1.f);
  }
}

__global__ void count_nodes_kernel(const int* __restrict__ node_gid,
                                   float* __restrict__ gN_count) {
  int i = blockIdx.x * blockDim.x + threadIdx.x;
  if (i < NNODES) atomicAdd(&gN_count[node_gid[i]], 1.f);
}

// Pack W (din x 16*ntiles, row-major f32) into WMMA B-operand layout, f16.
// B layout (mirror of documented 16-bit A layout): lane l holds column
// n = l&15, kb = (l>>4)*8, half j -> K = chunk*32 + kb + (j<8 ? j : j+8).
__global__ void pack_b_kernel(const float* __restrict__ W, half_t* __restrict__ out,
                              int din, int ntiles, int nchunks) {
  int tid = blockIdx.x * blockDim.x + threadIdx.x;
  int total = ntiles * nchunks * 512;
  if (tid >= total) return;
  int j = tid & 15;
  int l = (tid >> 4) & 31;
  int c = (tid >> 9) % nchunks;
  int t = tid / (512 * nchunks);
  int n  = l & 15;
  int kb = (l >> 4) * 8;
  int k  = c * 32 + kb + (j < 8 ? j : j + 8);
  float v = (k < din) ? W[k * (16 * ntiles) + t * 16 + n] : 0.f;
  out[tid] = (half_t)v;  // tid == ((t*nchunks + c)*32 + l)*16 + j
}

// ---------------------------------------------------------------------------
// Edge block: one wave per 16-edge tile, 4 waves per block.
// e_in = [edge(32) | nodes[recv](32) | nodes[send](32) | glob[gid](16 pad 32)]
// MLP 112->16 (4 WMMA) ->16 (1 WMMA) ->32 (2 WMMA), relu, relu.
// Writes edges in place (own rows only); scatters into node/graph sums.
__launch_bounds__(128)
__global__ void edge_mlp_kernel(
    const float* __restrict__ nodes, const float* __restrict__ edges_in,
    float* __restrict__ edges_out, const float* __restrict__ glob,
    const int* __restrict__ senders, const int* __restrict__ receivers,
    const int* __restrict__ edge_gid,
    const v16h* __restrict__ w1p, const v16h* __restrict__ w2p,
    const v16h* __restrict__ w3p,
    const float* __restrict__ b1, const float* __restrict__ b2,
    const float* __restrict__ b3,
    float* __restrict__ node_accum, float* __restrict__ gE_accum)
{
  __shared__ half_t stage[4][2][256];
  const int lane = threadIdx.x & 31;
  const int wave = threadIdx.x >> 5;
  const int tile = blockIdx.x * 4 + wave;
  if (tile * 16 >= NEDGES) return;  // uniform per wave; grid sized exactly
  const int e0 = tile * 16;
  const int m  = lane & 15;
  const int kb = (lane >> 4) * 8;
  const int hi = (lane >> 4) * 8;
  const int e  = e0 + m;
  const int rcv = receivers[e];
  const int snd = senders[e];
  const int gid = edge_gid[e];
  const float* erow = edges_in + (size_t)e   * 32;
  const float* rrow = nodes    + (size_t)rcv * 32;
  const float* srow = nodes    + (size_t)snd * 32;
  const float* grow = glob     + (size_t)gid * 16;

  // ---- layer 1: 112 -> 16, K chunks {edge, recv, send, glob(pad)} ----
  v16h a0, a1, a2, a3;
#pragma unroll
  for (int j = 0; j < 16; ++j) {
    int k = kb + (j < 8 ? j : j + 8);          // A layout K within chunk
    a0[j] = (half_t)erow[k];
    a1[j] = (half_t)rrow[k];
    a2[j] = (half_t)srow[k];
    a3[j] = (j < 8) ? (half_t)grow[kb + j] : (half_t)0.f;
  }
  v8f acc = {};
  acc = __builtin_amdgcn_wmma_f32_16x16x32_f16(false, a0, false, w1p[0 * 32 + lane], (short)0, acc, false, false);
  acc = __builtin_amdgcn_wmma_f32_16x16x32_f16(false, a1, false, w1p[1 * 32 + lane], (short)0, acc, false, false);
  acc = __builtin_amdgcn_wmma_f32_16x16x32_f16(false, a2, false, w1p[2 * 32 + lane], (short)0, acc, false, false);
  acc = __builtin_amdgcn_wmma_f32_16x16x32_f16(false, a3, false, w1p[3 * 32 + lane], (short)0, acc, false, false);

  {  // bias + relu, stage D (col-striped) -> LDS [M][N]
    float bias = b1[m];
    half_t* hb = stage[wave][0];
#pragma unroll
    for (int r = 0; r < 8; ++r)
      hb[(r + hi) * 16 + m] = (half_t)fmaxf(acc[r] + bias, 0.f);
  }
  __syncthreads();

  // ---- layer 2: 16 -> 16 (K padded to 32 with zeros) ----
  v16h a;
  {
    const half_t* hb = stage[wave][0];
#pragma unroll
    for (int j = 0; j < 8; ++j) a[j] = hb[m * 16 + kb + j];
#pragma unroll
    for (int j = 8; j < 16; ++j) a[j] = (half_t)0.f;
  }
  v8f h2 = {};
  h2 = __builtin_amdgcn_wmma_f32_16x16x32_f16(false, a, false, w2p[lane], (short)0, h2, false, false);
  {
    float bias = b2[m];
    half_t* hb = stage[wave][1];
#pragma unroll
    for (int r = 0; r < 8; ++r)
      hb[(r + hi) * 16 + m] = (half_t)fmaxf(h2[r] + bias, 0.f);
  }
  __syncthreads();

  // ---- layer 3: 16 -> 32 (two N tiles) ----
  {
    const half_t* hb = stage[wave][1];
#pragma unroll
    for (int j = 0; j < 8; ++j) a[j] = hb[m * 16 + kb + j];  // a[8..15] stay 0
  }
  v8f d0 = {}, d1 = {};
  d0 = __builtin_amdgcn_wmma_f32_16x16x32_f16(false, a, false, w3p[0 * 32 + lane], (short)0, d0, false, false);
  d1 = __builtin_amdgcn_wmma_f32_16x16x32_f16(false, a, false, w3p[1 * 32 + lane], (short)0, d1, false, false);

  const float bo0 = b3[m], bo1 = b3[16 + m];
#pragma unroll
  for (int r = 0; r < 8; ++r) {
    int row = r + hi;
    int ee  = e0 + row;
    float v0 = d0[r] + bo0;
    float v1 = d1[r] + bo1;
    edges_out[(size_t)ee * 32 + m]      = v0;
    edges_out[(size_t)ee * 32 + 16 + m] = v1;
    int rr = receivers[ee];
    int gg = edge_gid[ee];
    atomicAdd(&node_accum[(size_t)rr * 32 + m],      v0);
    atomicAdd(&node_accum[(size_t)rr * 32 + 16 + m], v1);
    atomicAdd(&gE_accum[gg * 32 + m],      v0);
    atomicAdd(&gE_accum[gg * 32 + 16 + m], v1);
  }
}

// ---------------------------------------------------------------------------
// Node block: one wave per 16-node tile, 2 waves per block.
// n_in = [agg_e(32) | node(32) | glob[gid](16 pad 32)], MLP 80->16->16->32.
__launch_bounds__(64)
__global__ void node_mlp_kernel(
    const float* __restrict__ node_accum, const float* __restrict__ node_count,
    const float* __restrict__ nodes_in, float* __restrict__ nodes_out,
    const float* __restrict__ glob, const int* __restrict__ node_gid,
    const v16h* __restrict__ w1p, const v16h* __restrict__ w2p,
    const v16h* __restrict__ w3p,
    const float* __restrict__ b1, const float* __restrict__ b2,
    const float* __restrict__ b3,
    float* __restrict__ gN_accum)
{
  __shared__ half_t stage[2][2][256];
  const int lane = threadIdx.x & 31;
  const int wave = threadIdx.x >> 5;
  const int tile = blockIdx.x * 2 + wave;
  if (tile * 16 >= NNODES) return;  // uniform per wave; grid sized exactly
  const int i0 = tile * 16;
  const int m  = lane & 15;
  const int kb = (lane >> 4) * 8;
  const int hi = (lane >> 4) * 8;
  const int i  = i0 + m;
  const int gid = node_gid[i];
  const float invc = 1.f / fmaxf(node_count[i], 1.f);
  const float* arow = node_accum + (size_t)i * 32;
  const float* nrow = nodes_in   + (size_t)i * 32;
  const float* grow = glob       + (size_t)gid * 16;

  v16h a0, a1, a2;
#pragma unroll
  for (int j = 0; j < 16; ++j) {
    int k = kb + (j < 8 ? j : j + 8);
    a0[j] = (half_t)(arow[k] * invc);   // segment mean of received edges
    a1[j] = (half_t)nrow[k];
    a2[j] = (j < 8) ? (half_t)grow[kb + j] : (half_t)0.f;
  }
  v8f acc = {};
  acc = __builtin_amdgcn_wmma_f32_16x16x32_f16(false, a0, false, w1p[0 * 32 + lane], (short)0, acc, false, false);
  acc = __builtin_amdgcn_wmma_f32_16x16x32_f16(false, a1, false, w1p[1 * 32 + lane], (short)0, acc, false, false);
  acc = __builtin_amdgcn_wmma_f32_16x16x32_f16(false, a2, false, w1p[2 * 32 + lane], (short)0, acc, false, false);

  {
    float bias = b1[m];
    half_t* hb = stage[wave][0];
#pragma unroll
    for (int r = 0; r < 8; ++r)
      hb[(r + hi) * 16 + m] = (half_t)fmaxf(acc[r] + bias, 0.f);
  }
  __syncthreads();

  v16h a;
  {
    const half_t* hb = stage[wave][0];
#pragma unroll
    for (int j = 0; j < 8; ++j) a[j] = hb[m * 16 + kb + j];
#pragma unroll
    for (int j = 8; j < 16; ++j) a[j] = (half_t)0.f;
  }
  v8f h2 = {};
  h2 = __builtin_amdgcn_wmma_f32_16x16x32_f16(false, a, false, w2p[lane], (short)0, h2, false, false);
  {
    float bias = b2[m];
    half_t* hb = stage[wave][1];
#pragma unroll
    for (int r = 0; r < 8; ++r)
      hb[(r + hi) * 16 + m] = (half_t)fmaxf(h2[r] + bias, 0.f);
  }
  __syncthreads();

  {
    const half_t* hb = stage[wave][1];
#pragma unroll
    for (int j = 0; j < 8; ++j) a[j] = hb[m * 16 + kb + j];
  }
  v8f d0 = {}, d1 = {};
  d0 = __builtin_amdgcn_wmma_f32_16x16x32_f16(false, a, false, w3p[0 * 32 + lane], (short)0, d0, false, false);
  d1 = __builtin_amdgcn_wmma_f32_16x16x32_f16(false, a, false, w3p[1 * 32 + lane], (short)0, d1, false, false);

  const float bo0 = b3[m], bo1 = b3[16 + m];
#pragma unroll
  for (int r = 0; r < 8; ++r) {
    int row = r + hi;
    int ii  = i0 + row;
    float v0 = d0[r] + bo0;
    float v1 = d1[r] + bo1;
    nodes_out[(size_t)ii * 32 + m]      = v0;
    nodes_out[(size_t)ii * 32 + 16 + m] = v1;
    int gg = node_gid[ii];
    atomicAdd(&gN_accum[gg * 32 + m],      v0);
    atomicAdd(&gN_accum[gg * 32 + 16 + m], v1);
  }
}

// ---------------------------------------------------------------------------
// Global block: 64 graphs, one thread each (trivial FLOPs; f32 VALU).
__launch_bounds__(64)
__global__ void global_mlp_kernel(
    const float* __restrict__ gE_accum, const float* __restrict__ gE_count,
    const float* __restrict__ gN_accum, const float* __restrict__ gN_count,
    const float* __restrict__ glob_in, float* __restrict__ glob_out,
    const float* __restrict__ w1, const float* __restrict__ b1,
    const float* __restrict__ w2, const float* __restrict__ b2,
    const float* __restrict__ w3, const float* __restrict__ b3)
{
  int g = threadIdx.x;
  if (g >= NGRAPH) return;
  float x[80];
  float ie = 1.f / fmaxf(gE_count[g], 1.f);
  float in_ = 1.f / fmaxf(gN_count[g], 1.f);
  for (int c = 0; c < 32; ++c) x[c]      = gE_accum[g * 32 + c] * ie;
  for (int c = 0; c < 32; ++c) x[32 + c] = gN_accum[g * 32 + c] * in_;
  for (int c = 0; c < 16; ++c) x[64 + c] = glob_in[g * 16 + c];
  float h1[16], h2[16];
  for (int n = 0; n < 16; ++n) {
    float s = b1[n];
    for (int k = 0; k < 80; ++k) s += x[k] * w1[k * 16 + n];
    h1[n] = fmaxf(s, 0.f);
  }
  for (int n = 0; n < 16; ++n) {
    float s = b2[n];
    for (int k = 0; k < 16; ++k) s += h1[k] * w2[k * 16 + n];
    h2[n] = fmaxf(s, 0.f);
  }
  for (int n = 0; n < 16; ++n) {
    float s = b3[n];
    for (int k = 0; k < 16; ++k) s += h2[k] * w3[k * 16 + n];
    glob_out[g * 16 + n] = s;
  }
}

// ---------------------------------------------------------------------------
extern "C" void kernel_launch(void* const* d_in, const int* in_sizes, int n_in,
                              void* d_out, int out_size, void* d_ws, size_t ws_size,
                              hipStream_t stream) {
  (void)in_sizes; (void)n_in; (void)out_size; (void)ws_size;
  const float* in_nodes = (const float*)d_in[0];
  const float* in_edges = (const float*)d_in[1];
  const float* in_glob  = (const float*)d_in[2];
  const int* senders    = (const int*)d_in[3];
  const int* receivers  = (const int*)d_in[4];
  const int* node_gid   = (const int*)d_in[5];
  const int* edge_gid   = (const int*)d_in[6];
  const float* e_w1 = (const float*)d_in[7];  const float* e_b1 = (const float*)d_in[8];
  const float* e_w2 = (const float*)d_in[9];  const float* e_b2 = (const float*)d_in[10];
  const float* e_w3 = (const float*)d_in[11]; const float* e_b3 = (const float*)d_in[12];
  const float* n_w1 = (const float*)d_in[13]; const float* n_b1 = (const float*)d_in[14];
  const float* n_w2 = (const float*)d_in[15]; const float* n_b2 = (const float*)d_in[16];
  const float* n_w3 = (const float*)d_in[17]; const float* n_b3 = (const float*)d_in[18];
  const float* g_w1 = (const float*)d_in[19]; const float* g_b1 = (const float*)d_in[20];
  const float* g_w2 = (const float*)d_in[21]; const float* g_b2 = (const float*)d_in[22];
  const float* g_w3 = (const float*)d_in[23]; const float* g_b3 = (const float*)d_in[24];

  float* out_nodes = (float*)d_out;                       // 100000*32
  float* out_edges = out_nodes + (size_t)NNODES * 32;     // 800000*32
  float* out_glob  = out_edges + (size_t)NEDGES * 32;     // 64*16

  // ---- scratch carve (≈13.5 MB) ----
  char* base = (char*)d_ws;
  size_t off = 0;
  auto carve = [&](size_t bytes) -> void* {
    void* p = base + off;
    off += (bytes + 255) & ~(size_t)255;
    return p;
  };
  float* node_accum = (float*)carve((size_t)NNODES * 32 * 4);
  float* node_count = (float*)carve((size_t)NNODES * 4);
  float* gE_accum   = (float*)carve(NGRAPH * 32 * 4);
  float* gE_count   = (float*)carve(NGRAPH * 4);
  float* gN_accum   = (float*)carve(NGRAPH * 32 * 4);
  float* gN_count   = (float*)carve(NGRAPH * 4);
  half_t* ew1p = (half_t*)carve(4 * 512 * 2);
  half_t* ew2p = (half_t*)carve(1 * 512 * 2);
  half_t* ew3p = (half_t*)carve(2 * 512 * 2);
  half_t* nw1p = (half_t*)carve(3 * 512 * 2);
  half_t* nw2p = (half_t*)carve(1 * 512 * 2);
  half_t* nw3p = (half_t*)carve(2 * 512 * 2);

  // ---- pack weights to WMMA B layout (tiny, once per launch) ----
  pack_b_kernel<<<8, 256, 0, stream>>>(e_w1, ew1p, 112, 1, 4);
  pack_b_kernel<<<2, 256, 0, stream>>>(e_w2, ew2p, 16, 1, 1);
  pack_b_kernel<<<4, 256, 0, stream>>>(e_w3, ew3p, 16, 2, 1);
  pack_b_kernel<<<6, 256, 0, stream>>>(n_w1, nw1p, 80, 1, 3);
  pack_b_kernel<<<2, 256, 0, stream>>>(n_w2, nw2p, 16, 1, 1);
  pack_b_kernel<<<4, 256, 0, stream>>>(n_w3, nw3p, 16, 2, 1);

  // ---- segment counts (constant across passes) ----
  zero_kernel<<<(NNODES + 255) / 256, 256, 0, stream>>>(node_count, NNODES);
  zero_kernel<<<1, 256, 0, stream>>>(gE_count, NGRAPH);
  zero_kernel<<<1, 256, 0, stream>>>(gN_count, NGRAPH);
  count_edges_kernel<<<(NEDGES + 255) / 256, 256, 0, stream>>>(receivers, edge_gid,
                                                               node_count, gE_count);
  count_nodes_kernel<<<(NNODES + 255) / 256, 256, 0, stream>>>(node_gid, gN_count);

  for (int p = 0; p < NPASS; ++p) {
    const float* nodes_in = (p == 0) ? in_nodes : out_nodes;
    const float* edges_in = (p == 0) ? in_edges : out_edges;
    const float* glob_in  = (p == 0) ? in_glob  : out_glob;

    zero_kernel<<<(NNODES * 32 + 255) / 256, 256, 0, stream>>>(node_accum, NNODES * 32);
    zero_kernel<<<(NGRAPH * 32 + 255) / 256, 256, 0, stream>>>(gE_accum, NGRAPH * 32);
    zero_kernel<<<(NGRAPH * 32 + 255) / 256, 256, 0, stream>>>(gN_accum, NGRAPH * 32);

    // 50000 tiles of 16 edges, 4 waves/block
    edge_mlp_kernel<<<NEDGES / 16 / 4, 128, 0, stream>>>(
        nodes_in, edges_in, out_edges, glob_in,
        senders, receivers, edge_gid,
        (const v16h*)ew1p, (const v16h*)ew2p, (const v16h*)ew3p,
        e_b1, e_b2, e_b3, node_accum, gE_accum);

    // 6250 tiles of 16 nodes, 2 waves/block
    node_mlp_kernel<<<NNODES / 16 / 2, 64, 0, stream>>>(
        node_accum, node_count, nodes_in, out_nodes, glob_in, node_gid,
        (const v16h*)nw1p, (const v16h*)nw2p, (const v16h*)nw3p,
        n_b1, n_b2, n_b3, gN_accum);

    global_mlp_kernel<<<1, 64, 0, stream>>>(
        gE_accum, gE_count, gN_accum, gN_count, glob_in, out_glob,
        g_w1, g_b1, g_w2, g_b2, g_w3, g_b3);
  }
}
